// InstrumentedSPAIGenerator_88948772700942
// MI455X (gfx1250) — matrix operand order repaired and checked by reading, not compiled, not run on hardware
//
#include <hip/hip_runtime.h>
#include <hip/hip_bf16.h>
#include <math.h>

// ---------------------------------------------------------------------------
// Fully fused windowed transformer for MI455X (gfx1250, wave32, WMMA).
// One workgroup (256 threads = 8 waves) per 256-token window; ~310 KB LDS.
// All GEMMs (embedding proj, QKV, scores, A*V, out-proj, FF1, FF2, head) run
// on v_wmma_f32_16x16x32_f16. WMMA operands are staged in LDS as f16 in
// fragment-friendly layouts: every A/B fragment = 2x ds_load_b128 per lane.
// ALL model weights (4 layers + embed + head, 70 KB f16) are staged into LDS
// once at kernel start — no per-layer staging stalls.
// ---------------------------------------------------------------------------

typedef __attribute__((ext_vector_type(16))) _Float16 v16h;
typedef __attribute__((ext_vector_type(8)))  _Float16 v8h;
typedef __attribute__((ext_vector_type(8)))  float    v8f;

#define WIN    256
#define DM     32
#define NH     4
#define DH     8
#define DFF    64
#define NLAYER 4
#define INDIM  58
#define HOUT   25

__device__ __forceinline__ v8f wmma32(v16h a, v16h b, v8f c) {
  return __builtin_amdgcn_wmma_f32_16x16x32_f16(
      false, a, false, b, (short)0, c, false, false);
}

__device__ __forceinline__ v16h cat16(v8h lo, v8h hi) {
  v16h r;
#pragma unroll
  for (int e = 0; e < 8; ++e) { r[e] = lo[e]; r[e + 8] = hi[e]; }
  return r;
}

__device__ __forceinline__ v16h zero16() {
  v16h z;
#pragma unroll
  for (int e = 0; e < 16; ++e) z[e] = (_Float16)0.0f;
  return z;
}

// A-fragment (16x32 f16) from row-major f16 [.., stride] starting at row0.
// ISA 7.12.2: lanes 0-15: M=lane, K {0..7,16..23}; lanes 16-31: M=lane-16,
// K {8..15,24..31}.  -> two 16B LDS loads per lane.
__device__ __forceinline__ v16h load_a16(const _Float16* base, int stride, int lane) {
  const int row = lane & 15;
  const int kb1 = (lane & 16) ? 8 : 0;
  const _Float16* p = base + row * stride + kb1;
  return cat16(*(const v8h*)p, *(const v8h*)(p + 16));
}

// B-fragment (32x16 f16) from n-major f16 (row index = n, k contiguous).
// lanes 0-15 hold K {0..7,16..23}, lanes 16-31 hold K {8..15,24..31}, N = lane%16.
__device__ __forceinline__ v16h load_b16(const _Float16* base, int stride, int lane) {
  const int n  = lane & 15;
  const int kb = (lane & 16) ? 8 : 0;
  const _Float16* p = base + n * stride + kb;
  return cat16(*(const v8h*)p, *(const v8h*)(p + 16));
}

// C/D fragment: VGPR r -> row r (lanes 0-15) or 8+r (lanes 16-31), N=lane%16.
template <typename F>
__device__ __forceinline__ void store_d_frag(int lane, v8f d, F f) { // f(m,n,v)
  const int n  = lane & 15;
  const int mb = (lane & 16) ? 8 : 0;
#pragma unroll
  for (int r = 0; r < 8; ++r) f(mb + r, n, d[r]);
}

__global__ __launch_bounds__(256) void spai_fused_transformer(
    const float* __restrict__ x,       const int*   __restrict__ layers,
    const float* __restrict__ fpw,     const float* __restrict__ fpb,
    const float* __restrict__ lemb,    const float* __restrict__ fourB,
    const float* __restrict__ ipw_all, const float* __restrict__ ipb_all,
    const float* __restrict__ opw_all, const float* __restrict__ opb_all,
    const float* __restrict__ ln1g_a,  const float* __restrict__ ln1b_a,
    const float* __restrict__ w1_all,  const float* __restrict__ bf1_all,
    const float* __restrict__ w2_all,  const float* __restrict__ bf2_all,
    const float* __restrict__ ln2g_a,  const float* __restrict__ ln2b_a,
    const float* __restrict__ nog,     const float* __restrict__ nob,
    const float* __restrict__ hw,      const float* __restrict__ hb,
    float* __restrict__ out)
{
  extern __shared__ __align__(16) char smem[];
  float*    sh_h   = (float*)smem;                          //      0  32 KB f32 residual
  _Float16* sh_hh  = (_Float16*)(smem + 32768);             //  32768  16 KB f16 copy of h
  _Float16* sh_qk  = (_Float16*)(smem + 49152);             //  49152  32 KB [256][64] q|k
  _Float16* sh_vt  = (_Float16*)(smem + 81920);             //  81920  16 KB [32][256] v^T
  _Float16* sh_att = (_Float16*)(smem + 98304);             //  98304  16 KB [256][32]
  _Float16* sh_probs = (_Float16*)(smem + 114688);          // 114688 128 KB [256][256]
  float*    sh_scr = (float*)(smem + 114688);               // alias: [256][32] f32
  _Float16* sh_ff  = (_Float16*)(smem + 114688 + 32768);    // alias: [256][64] f16
  _Float16* sh_x16 = (_Float16*)(smem + 114688);            // alias: [256][64] f16 (embed)
  _Float16* sh_w   = (_Float16*)(smem + 245760);            // 245760  70 KB all weights f16
  // sh_w layout (halves): [0..2047] fpw16 [32][64];  [2048..3071] hw16 [32][32];
  //   per layer li at 3072 + li*8192: wq16[96][32], wo16[32][32], w116[64][32], w216[32][64]

  const int  tid  = threadIdx.x;
  const int  wv   = tid >> 5;
  const int  lane = tid & 31;
  const long tok0 = (long)blockIdx.x * WIN;

  // ============== stage ALL weights (f16) + x (f16) into LDS once ===========
  for (int idx = tid; idx < 2048; idx += 256) {            // fpw [32][64], K pad 58->64
    const int r = idx >> 6, c = idx & 63;
    sh_w[idx] = (_Float16)((c < INDIM) ? fpw[r * INDIM + c] : 0.0f);
  }
  for (int idx = tid; idx < 1024; idx += 256) {            // head_w [32][32], rows>=25 zero
    const int r = idx >> 5, c = idx & 31;
    sh_w[2048 + idx] = (_Float16)((r < HOUT) ? hw[r * DM + c] : 0.0f);
  }
  for (int idx = tid; idx < NLAYER * 3072; idx += 256) {   // in_proj [96][32] per layer
    const int li = idx / 3072;
    sh_w[3072 + li * 8192 + (idx - li * 3072)] = (_Float16)ipw_all[idx];
  }
  for (int idx = tid; idx < NLAYER * 1024; idx += 256) {   // out_proj [32][32]
    const int li = idx >> 10;
    sh_w[3072 + li * 8192 + 3072 + (idx & 1023)] = (_Float16)opw_all[idx];
  }
  for (int idx = tid; idx < NLAYER * 2048; idx += 256) {   // lin1 [64][32]
    const int li = idx >> 11;
    sh_w[3072 + li * 8192 + 4096 + (idx & 2047)] = (_Float16)w1_all[idx];
  }
  for (int idx = tid; idx < NLAYER * 2048; idx += 256) {   // lin2 [32][64]
    const int li = idx >> 11;
    sh_w[3072 + li * 8192 + 6144 + (idx & 2047)] = (_Float16)w2_all[idx];
  }
  {
    const long g = tok0 + tid;
    const float* xr = x + g * INDIM;
    _Float16* xp = sh_x16 + tid * 64;
#pragma unroll
    for (int i = 0; i < INDIM; ++i) xp[i] = (_Float16)xr[i];
#pragma unroll
    for (int i = INDIM; i < 64; ++i) xp[i] = (_Float16)0.0f;
  }
  __syncthreads();

  // ===================== embedding: h = x @ Wf^T (WMMA) =====================
  for (int t = wv; t < 32; t += 8) {   // 16 row-tiles x 2 col-tiles, K=64
    const int rt = t >> 1, ct = t & 1;
    v8f acc = {};
    for (int kb = 0; kb < 2; ++kb) {
      v16h A = load_a16(sh_x16 + (rt * 16) * 64 + kb * 32, 64, lane);
      v16h B = load_b16(sh_w + (ct * 16) * 64 + kb * 32, 64, lane);
      acc = wmma32(A, B, acc);
    }
    store_d_frag(lane, acc, [&](int m, int n, float v) {
      sh_h[(rt * 16 + m) * DM + ct * 16 + n] = v;
    });
  }
  __syncthreads();
  // finalize embedding per token: + bias + layer_embed + fourier PE.
  // one phase per dd serves both sin (d=dd) and cos (d=dd+16).
  {
    const int  s = tid;
    const long g = tok0 + s;
    const float p0 = x[g * INDIM + 0], p1 = x[g * INDIM + 1], p2 = x[g * INDIM + 2];
    const int  ly = layers[g];
    const float TWO_PI = 6.283185307179586f;
    float val[DM];
#pragma unroll
    for (int dd = 0; dd < 16; ++dd) {
      const float pr = TWO_PI * (p0 * fourB[dd] + p1 * fourB[16 + dd] +
                                 p2 * fourB[32 + dd]);
      val[dd]      = sh_h[s * DM + dd]      + fpb[dd]      + lemb[ly * DM + dd]      + __sinf(pr);
      val[dd + 16] = sh_h[s * DM + dd + 16] + fpb[dd + 16] + lemb[ly * DM + dd + 16] + __cosf(pr);
    }
#pragma unroll
    for (int d = 0; d < DM; ++d) sh_h[s * DM + d] = val[d];
    v8h* hp = (v8h*)(sh_hh + s * DM);
#pragma unroll
    for (int q8 = 0; q8 < 4; ++q8) {
      v8h t;
#pragma unroll
      for (int e = 0; e < 8; ++e) t[e] = (_Float16)val[q8 * 8 + e];
      hp[q8] = t;
    }
  }
  __syncthreads();

  // ================================ layers ==================================
  const float iss = 0.3535533905932738f;  // 1/sqrt(dh), folded into q
  for (int li = 0; li < NLAYER; ++li) {
    const float* ipb = ipb_all + li * 3 * DM;
    const float* opb = opb_all + li * DM;
    const float* g1  = ln1g_a + li * DM;
    const float* b1  = ln1b_a + li * DM;
    const float* bf1 = bf1_all + li * DFF;
    const float* bf2 = bf2_all + li * DM;
    const float* g2  = ln2g_a + li * DM;
    const float* b2  = ln2b_a + li * DM;
    const _Float16* wq16 = sh_w + 3072 + li * 8192;          // [96][32]
    const _Float16* wo16 = wq16 + 3072;                      // [32][32]
    const _Float16* w116 = wq16 + 4096;                      // [64][32]
    const _Float16* w216 = wq16 + 6144;                      // [32][64]

    // ---- QKV: [256,32]@[32,96]; q,k row-major f16; v stored transposed -----
    for (int t = wv; t < 96; t += 8) {
      const int rt = t / 6, ct = t % 6;
      v16h A = load_a16(sh_hh + (rt * 16) * DM, DM, lane);
      v16h B = load_b16(wq16 + (ct * 16) * DM, DM, lane);
      v8f C = {};
      C = wmma32(A, B, C);
      if (ct < 4) {  // q (scaled by 1/sqrt(dh)) and k -> sh_qk [256][64]
        const float sc = (ct < 2) ? iss : 1.0f;
        store_d_frag(lane, C, [&](int m, int n, float v) {
          const int col = ct * 16 + n;
          sh_qk[(rt * 16 + m) * 64 + col] = (_Float16)((v + ipb[col]) * sc);
        });
      } else {       // v -> sh_vt [32][256] transposed
        store_d_frag(lane, C, [&](int m, int n, float v) {
          const int col = ct * 16 + n;   // 64..95
          sh_vt[(col - 64) * WIN + rt * 16 + m] = (_Float16)(v + ipb[col]);
        });
      }
    }
    __syncthreads();

    // ---- attention (heads sequential; probs buffer reused per head) --------
    for (int hd = 0; hd < NH; ++hd) {
      for (int rr = 0; rr < 2; ++rr) {     // 16 row-tiles, 2 per wave
        const int rt = wv * 2 + rr;
        // A = q rows (dh=8, K padded to 32): lanes>=16 carry zeros
        v16h Aq = zero16();
        if (!(lane & 16)) {
          const v8h qv = *(const v8h*)(sh_qk + ((rt * 16) + (lane & 15)) * 64 + hd * DH);
#pragma unroll
          for (int e = 0; e < 8; ++e) Aq[e] = qv[e];
        }
        v8f sc[16];
        for (int ct = 0; ct < 16; ++ct) {
          v16h Bk = zero16();
          if (!(lane & 16)) {
            const v8h kv = *(const v8h*)(sh_qk + ((ct * 16) + (lane & 15)) * 64 + DM + hd * DH);
#pragma unroll
            for (int e = 0; e < 8; ++e) Bk[e] = kv[e];
          }
          v8f z = {};
          sc[ct] = wmma32(Aq, Bk, z);    // already scaled via q
        }
        float rdiv[8];
#pragma unroll
        for (int r = 0; r < 8; ++r) {
          float mx = -3.4e38f;
#pragma unroll
          for (int ct = 0; ct < 16; ++ct) mx = fmaxf(mx, sc[ct][r]);
          mx = fmaxf(mx, __shfl_xor(mx, 1));
          mx = fmaxf(mx, __shfl_xor(mx, 2));
          mx = fmaxf(mx, __shfl_xor(mx, 4));
          mx = fmaxf(mx, __shfl_xor(mx, 8));
          float sum = 0.0f;
#pragma unroll
          for (int ct = 0; ct < 16; ++ct) {
            const float e = __expf(sc[ct][r] - mx);
            sc[ct][r] = e;
            sum += e;
          }
          sum += __shfl_xor(sum, 1);
          sum += __shfl_xor(sum, 2);
          sum += __shfl_xor(sum, 4);
          sum += __shfl_xor(sum, 8);
          rdiv[r] = 1.0f / sum;
        }
        const int n  = lane & 15;
        const int mb = (lane & 16) ? 8 : 0;
#pragma unroll
        for (int ct = 0; ct < 16; ++ct)
#pragma unroll
          for (int r = 0; r < 8; ++r)
            sh_probs[(rt * 16 + mb + r) * WIN + ct * 16 + n] =
                (_Float16)(sc[ct][r] * rdiv[r]);
      }
      __syncthreads();

      // probs @ V : K=256 -> 8 accumulating WMMAs; B from transposed V
      for (int rr = 0; rr < 2; ++rr) {
        const int rt = wv * 2 + rr;
        v8f acc = {};
        const int n2  = lane & 15;
        const int kbh = (lane & 16) ? 8 : 0;
        for (int kb = 0; kb < 8; ++kb) {
          v16h Ap = load_a16(sh_probs + (rt * 16) * WIN + kb * 32, WIN, lane);
          v16h Bv = zero16();
          if (n2 < DH) {
            const _Float16* p = sh_vt + (hd * DH + n2) * WIN + kb * 32 + kbh;
            const v8h lo = *(const v8h*)p;
            const v8h hi = *(const v8h*)(p + 16);
#pragma unroll
            for (int e = 0; e < 8; ++e) { Bv[e] = lo[e]; Bv[e + 8] = hi[e]; }
          }
          acc = wmma32(Ap, Bv, acc);
        }
        store_d_frag(lane, acc, [&](int m, int n3, float v) {
          if (n3 < DH) sh_att[(rt * 16 + m) * DM + hd * DH + n3] = (_Float16)v;
        });
      }
      __syncthreads();  // probs reused by the next head
    }

    // ---- out-projection: [256,32]@[32,32] ----------------------------------
    for (int t = wv; t < 32; t += 8) {
      const int rt = t >> 1, ct = t & 1;
      v16h A = load_a16(sh_att + (rt * 16) * DM, DM, lane);
      v16h B = load_b16(wo16 + (ct * 16) * DM, DM, lane);
      v8f C = {};
      C = wmma32(A, B, C);
      store_d_frag(lane, C, [&](int m, int n, float v) {
        sh_scr[(rt * 16 + m) * DM + ct * 16 + n] = v + opb[ct * 16 + n];
      });
    }
    __syncthreads();

    // ---- residual + LN1 (one token per thread); writes f32 h and f16 copy --
    {
      const int s = tid;
      float vbuf[DM];
      float mean = 0.0f;
#pragma unroll
      for (int d = 0; d < DM; ++d) {
        vbuf[d] = sh_h[s * DM + d] + sh_scr[s * DM + d];
        mean += vbuf[d];
      }
      mean *= (1.0f / DM);
      float var = 0.0f;
#pragma unroll
      for (int d = 0; d < DM; ++d) { const float t2 = vbuf[d] - mean; var += t2 * t2; }
      var *= (1.0f / DM);
      const float rstd = rsqrtf(var + 1e-5f);
      float o[DM];
#pragma unroll
      for (int d = 0; d < DM; ++d) {
        o[d] = (vbuf[d] - mean) * rstd * g1[d] + b1[d];
        sh_h[s * DM + d] = o[d];
      }
      v8h* hp = (v8h*)(sh_hh + s * DM);
#pragma unroll
      for (int q8 = 0; q8 < 4; ++q8) {
        v8h t;
#pragma unroll
        for (int e = 0; e < 8; ++e) t[e] = (_Float16)o[q8 * 8 + e];
        hp[q8] = t;
      }
    }
    __syncthreads();

    // ---- FF1 + exact GELU: [256,32]@[32,64] -> f16 [256][64] ---------------
    for (int t = wv; t < 64; t += 8) {
      const int rt = t >> 2, ct = t & 3;
      v16h A = load_a16(sh_hh + (rt * 16) * DM, DM, lane);
      v16h B = load_b16(w116 + (ct * 16) * DM, DM, lane);
      v8f C = {};
      C = wmma32(A, B, C);
      store_d_frag(lane, C, [&](int m, int n, float v) {
        const float u = v + bf1[ct * 16 + n];
        sh_ff[(rt * 16 + m) * DFF + ct * 16 + n] =
            (_Float16)(u * 0.5f * (1.0f + erff(u * 0.7071067811865476f)));
      });
    }
    __syncthreads();

    // ---- FF2: [256,64]@[64,32]; K=64 -> 2 WMMAs ----------------------------
    for (int t = wv; t < 32; t += 8) {
      const int rt = t >> 1, ct = t & 1;
      v8f acc = {};
      for (int kb = 0; kb < 2; ++kb) {
        v16h A = load_a16(sh_ff + (rt * 16) * DFF + kb * 32, DFF, lane);
        v16h B = load_b16(w216 + (ct * 16) * DFF + kb * 32, DFF, lane);
        acc = wmma32(A, B, acc);
      }
      store_d_frag(lane, acc, [&](int m, int n, float v) {
        sh_scr[(rt * 16 + m) * DM + ct * 16 + n] = v + bf2[ct * 16 + n];
      });
    }
    __syncthreads();

    // ---- residual + LN2 ----------------------------------------------------
    {
      const int s = tid;
      float vbuf[DM];
      float mean = 0.0f;
#pragma unroll
      for (int d = 0; d < DM; ++d) {
        vbuf[d] = sh_h[s * DM + d] + sh_scr[s * DM + d];
        mean += vbuf[d];
      }
      mean *= (1.0f / DM);
      float var = 0.0f;
#pragma unroll
      for (int d = 0; d < DM; ++d) { const float t2 = vbuf[d] - mean; var += t2 * t2; }
      var *= (1.0f / DM);
      const float rstd = rsqrtf(var + 1e-5f);
      float o[DM];
#pragma unroll
      for (int d = 0; d < DM; ++d) {
        o[d] = (vbuf[d] - mean) * rstd * g2[d] + b2[d];
        sh_h[s * DM + d] = o[d];
      }
      v8h* hp = (v8h*)(sh_hh + s * DM);
#pragma unroll
      for (int q8 = 0; q8 < 4; ++q8) {
        v8h t;
#pragma unroll
        for (int e = 0; e < 8; ++e) t[e] = (_Float16)o[q8 * 8 + e];
        hp[q8] = t;
      }
    }
    __syncthreads();
  } // layer loop

  // ================= final LN (-> f16) + head via WMMA ======================
  {
    const int s = tid;
    float vbuf[DM];
    float mean = 0.0f;
#pragma unroll
    for (int d = 0; d < DM; ++d) { vbuf[d] = sh_h[s * DM + d]; mean += vbuf[d]; }
    mean *= (1.0f / DM);
    float var = 0.0f;
#pragma unroll
    for (int d = 0; d < DM; ++d) { const float t2 = vbuf[d] - mean; var += t2 * t2; }
    var *= (1.0f / DM);
    const float rstd = rsqrtf(var + 1e-5f);
    v8h* hp = (v8h*)(sh_hh + s * DM);
#pragma unroll
    for (int q8 = 0; q8 < 4; ++q8) {
      v8h t;
#pragma unroll
      for (int e = 0; e < 8; ++e) {
        const int d = q8 * 8 + e;
        t[e] = (_Float16)((vbuf[d] - mean) * rstd * nog[d] + nob[d]);
      }
      hp[q8] = t;
    }
  }
  __syncthreads();
  for (int t = wv; t < 32; t += 8) {  // [256,32]@[32,32(25 used)]
    const int rt = t >> 1, ct = t & 1;
    v16h A = load_a16(sh_hh + (rt * 16) * DM, DM, lane);
    v16h B = load_b16(sh_w + 2048 + (ct * 16) * DM, DM, lane);
    v8f C = {};
    C = wmma32(A, B, C);
    store_d_frag(lane, C, [&](int m, int n, float v) {
      const int j = ct * 16 + n;
      if (j < HOUT)
        out[(tok0 + rt * 16 + m) * HOUT + j] = v + hb[j];
    });
  }
}

// ---------------------------------------------------------------------------
extern "C" void kernel_launch(void* const* d_in, const int* in_sizes, int n_in,
                              void* d_out, int out_size, void* d_ws, size_t ws_size,
                              hipStream_t stream) {
  (void)n_in; (void)out_size; (void)d_ws; (void)ws_size;
  const float* x    = (const float*)d_in[0];
  const int*   lay  = (const int*)d_in[1];
  const float* fpw  = (const float*)d_in[2];
  const float* fpb  = (const float*)d_in[3];
  const float* lemb = (const float*)d_in[4];
  const float* fB   = (const float*)d_in[5];
  const float* ipw  = (const float*)d_in[6];
  const float* ipb  = (const float*)d_in[7];
  const float* opw  = (const float*)d_in[8];
  const float* opb  = (const float*)d_in[9];
  const float* g1   = (const float*)d_in[10];
  const float* b1   = (const float*)d_in[11];
  const float* w1   = (const float*)d_in[12];
  const float* bf1  = (const float*)d_in[13];
  const float* w2   = (const float*)d_in[14];
  const float* bf2  = (const float*)d_in[15];
  const float* g2   = (const float*)d_in[16];
  const float* b2   = (const float*)d_in[17];
  const float* nog  = (const float*)d_in[18];
  const float* nob  = (const float*)d_in[19];
  const float* hw   = (const float*)d_in[20];
  const float* hb   = (const float*)d_in[21];
  float* out = (float*)d_out;

  const int tokens  = in_sizes[0] / INDIM;   // B*N = 131072
  const int nblocks = tokens / WIN;          // 512 windows

  // 240 KB activations + 70 KB whole-model f16 weights = 317440 B <= 320 KB
  const size_t smem_bytes = 245760 + 71680;

  hipFuncSetAttribute((const void*)spai_fused_transformer,
                      hipFuncAttributeMaxDynamicSharedMemorySize,
                      (int)smem_bytes);

  spai_fused_transformer<<<nblocks, 256, smem_bytes, stream>>>(
      x, lay, fpw, fpb, lemb, fB, ipw, ipb, opw, opb, g1, b1,
      w1, bf1, w2, bf2, g2, b2, nog, nob, hw, hb, out);
}
